// ModelVD_35253091565660
// MI455X (gfx1250) — matrix operand order repaired and checked by reading, not compiled, not run on hardware
//
#include <hip/hip_runtime.h>
#include <hip/hip_bf16.h>

// ---------------------------------------------------------------------------
// Types for CDNA5 WMMA
// ---------------------------------------------------------------------------
typedef __attribute__((ext_vector_type(16))) __bf16 v16bf;
typedef __attribute__((ext_vector_type(4)))  __bf16 v4bf;
typedef __attribute__((ext_vector_type(8)))  float  v8f;

#define BN_EPS 1e-5f

// ---------------------------------------------------------------------------
// WMMA fragment loaders (per CDNA5 ISA 7.12.2 VGPR layouts, wave32)
//   A 16x32 bf16:  lane = {half = lane>>4, m = lane&15}; element e:
//                  k = (e>>3)*16 + half*8 + (e&7)
//   B 32x16 bf16:  lane = {half, n = lane&15}; element e: k = (e>>3)*16 + half*8 + (e&7)
//   C/D 16x16 f32: VGPR r: row = r + 8*half, col = lane&15
// LDS layouts: lA[row][k] (stride LDL), lB[col][k] (stride LDL)
// ---------------------------------------------------------------------------
#define GBM 128
#define GBN 128
#define GBK 32
#define LDL 36          // 72B row stride: 8B-aligned packed stores, no bank dup
#define NT  256         // 8 waves

union FragU { v16bf v; unsigned u[8]; };

__device__ inline v16bf load_frag_a(const __bf16* lA, int rowBase, int lane) {
  const int half = lane >> 4, idx = lane & 15;
  const __bf16* base = lA + (rowBase + idx) * LDL;
  FragU f;
#pragma unroll
  for (int u = 0; u < 8; ++u) {
    const int g = u >> 2, w = (u & 3) * 2;
    f.u[u] = *(const unsigned*)(base + g * 16 + half * 8 + w);
  }
  return f.v;
}

__device__ inline v16bf load_frag_b(const __bf16* lBt, int colBase, int lane) {
  const int half = lane >> 4, idx = lane & 15;
  const __bf16* base = lBt + (colBase + idx) * LDL;
  FragU f;
#pragma unroll
  for (int u = 0; u < 8; ++u) {
    f.u[u] = *(const unsigned*)(base + (u >> 2) * 16 + half * 8 + (u & 3) * 2);
  }
  return f.v;
}

// ---------------------------------------------------------------------------
// Generic tiled GEMM: C[M,N] = act(A[M,K] @ op(B) + bias)
//   TRANSB == 0 : B is [K,N] row-major;  TRANSB == 1 : B is [N,K] row-major
// fp32 in/out, bf16 multiply via v_wmma_f32_16x16x32_bf16.
// 128x128 block tile, 256 threads (8 waves), wave = 32x64 sub-tile,
// K-step 32, register-prefetch software pipeline, peeled final iteration,
// block-uniform guard split (fast path = global_load_b128 + ds_store_b64).
// ---------------------------------------------------------------------------
template <int TRANSB>
__global__ __launch_bounds__(NT) void gemm_bf16_wmma(
    const float* __restrict__ A, const float* __restrict__ B,
    const float* __restrict__ bias, float* __restrict__ C,
    int M, int N, int K, int relu)
{
  __shared__ __align__(16) __bf16 lA[GBM * LDL];
  __shared__ __align__(16) __bf16 lB[GBN * LDL];

  const int tid  = threadIdx.x;
  const int lane = tid & 31;
  const int wave = tid >> 5;
  const int wm   = (wave >> 1) * 32;   // wave row offset (0..96)
  const int wn   = (wave & 1)  * 64;   // wave col offset (0,64)
  const int bm   = blockIdx.y * GBM;
  const int bn   = blockIdx.x * GBN;
  const bool fullMN = (bm + GBM <= M) && (bn + GBN <= N);

  float4 ra[4], rb[4];

  auto loadA = [&](int k0, bool guard) {
#pragma unroll
    for (int p = 0; p < 4; ++p) {
      const int l = p * 1024 + tid * 4;
      const int r = l >> 5, c = l & 31;
      const int gm = bm + r, gk = k0 + c;
      if (!guard) {
        ra[p] = *(const float4*)(A + (long)gm * K + gk);
      } else {
        float4 v = {0.f, 0.f, 0.f, 0.f};
        if (gm < M) {
          if (gk + 3 < K) v = *(const float4*)(A + (long)gm * K + gk);
          else {
            const float* src = A + (long)gm * K;
            if (gk + 0 < K) v.x = src[gk + 0];
            if (gk + 1 < K) v.y = src[gk + 1];
            if (gk + 2 < K) v.z = src[gk + 2];
            if (gk + 3 < K) v.w = src[gk + 3];
          }
        }
        ra[p] = v;
      }
    }
  };

  auto loadB = [&](int k0, bool guard) {
#pragma unroll
    for (int p = 0; p < 4; ++p) {
      const int l = p * 1024 + tid * 4;
      if (TRANSB) {
        const int n = l >> 5, c = l & 31;
        const int gn = bn + n, gk = k0 + c;
        if (!guard) {
          rb[p] = *(const float4*)(B + (long)gn * K + gk);
        } else {
          float4 v = {0.f, 0.f, 0.f, 0.f};
          if (gn < N) {
            if (gk + 3 < K) v = *(const float4*)(B + (long)gn * K + gk);
            else {
              const float* src = B + (long)gn * K;
              if (gk + 0 < K) v.x = src[gk + 0];
              if (gk + 1 < K) v.y = src[gk + 1];
              if (gk + 2 < K) v.z = src[gk + 2];
              if (gk + 3 < K) v.w = src[gk + 3];
            }
          }
          rb[p] = v;
        }
      } else {
        const int k = l >> 7, n = l & 127;
        const int gk = k0 + k, gn = bn + n;
        if (!guard) {
          rb[p] = *(const float4*)(B + (long)gk * N + gn);
        } else {
          float4 v = {0.f, 0.f, 0.f, 0.f};
          if (gk < K) {
            if (gn + 3 < N) v = *(const float4*)(B + (long)gk * N + gn);
            else {
              const float* src = B + (long)gk * N;
              if (gn + 0 < N) v.x = src[gn + 0];
              if (gn + 1 < N) v.y = src[gn + 1];
              if (gn + 2 < N) v.z = src[gn + 2];
              if (gn + 3 < N) v.w = src[gn + 3];
            }
          }
          rb[p] = v;
        }
      }
    }
  };

  auto storeA = [&]() {
#pragma unroll
    for (int p = 0; p < 4; ++p) {
      const int l = p * 1024 + tid * 4;
      const int r = l >> 5, c = l & 31;
      v4bf pk;
      pk.x = (__bf16)ra[p].x; pk.y = (__bf16)ra[p].y;
      pk.z = (__bf16)ra[p].z; pk.w = (__bf16)ra[p].w;
      *(v4bf*)(lA + r * LDL + c) = pk;
    }
  };

  auto storeB = [&]() {
#pragma unroll
    for (int p = 0; p < 4; ++p) {
      const int l = p * 1024 + tid * 4;
      if (TRANSB) {
        const int n = l >> 5, c = l & 31;
        v4bf pk;
        pk.x = (__bf16)rb[p].x; pk.y = (__bf16)rb[p].y;
        pk.z = (__bf16)rb[p].z; pk.w = (__bf16)rb[p].w;
        *(v4bf*)(lB + n * LDL + c) = pk;
      } else {
        const int k = l >> 7, n = l & 127;
        lB[(n + 0) * LDL + k] = (__bf16)rb[p].x;
        lB[(n + 1) * LDL + k] = (__bf16)rb[p].y;
        lB[(n + 2) * LDL + k] = (__bf16)rb[p].z;
        lB[(n + 3) * LDL + k] = (__bf16)rb[p].w;
      }
    }
  };

  const v8f vzero = {0.f, 0.f, 0.f, 0.f, 0.f, 0.f, 0.f, 0.f};
  v8f acc[2][4];
#pragma unroll
  for (int i = 0; i < 2; ++i)
#pragma unroll
    for (int j = 0; j < 4; ++j) acc[i][j] = vzero;

  // prologue: stage tile k0 = 0
  {
    const bool g = !fullMN || (GBK > K);
    loadA(0, g); loadB(0, g);
    storeA(); storeB();
  }
  __syncthreads();

  // steady state: always a next tile to prefetch (last iteration peeled)
  int k0 = 0;
  while (k0 + GBK < K) {
    v16bf fa0 = load_frag_a(lA, wm + 0,  lane);
    v16bf fa1 = load_frag_a(lA, wm + 16, lane);
    v16bf fb0 = load_frag_b(lB, wn + 0,  lane);
    v16bf fb1 = load_frag_b(lB, wn + 16, lane);
    v16bf fb2 = load_frag_b(lB, wn + 32, lane);
    v16bf fb3 = load_frag_b(lB, wn + 48, lane);

    const int kn = k0 + GBK;
    const bool g = !fullMN || (kn + GBK > K);
    loadA(kn, g); loadB(kn, g);     // overlaps with WMMAs below

    acc[0][0] = __builtin_amdgcn_wmma_f32_16x16x32_bf16(false, fa0, false, fb0, (short)0, acc[0][0], false, false);
    acc[0][1] = __builtin_amdgcn_wmma_f32_16x16x32_bf16(false, fa0, false, fb1, (short)0, acc[0][1], false, false);
    acc[0][2] = __builtin_amdgcn_wmma_f32_16x16x32_bf16(false, fa0, false, fb2, (short)0, acc[0][2], false, false);
    acc[0][3] = __builtin_amdgcn_wmma_f32_16x16x32_bf16(false, fa0, false, fb3, (short)0, acc[0][3], false, false);
    acc[1][0] = __builtin_amdgcn_wmma_f32_16x16x32_bf16(false, fa1, false, fb0, (short)0, acc[1][0], false, false);
    acc[1][1] = __builtin_amdgcn_wmma_f32_16x16x32_bf16(false, fa1, false, fb1, (short)0, acc[1][1], false, false);
    acc[1][2] = __builtin_amdgcn_wmma_f32_16x16x32_bf16(false, fa1, false, fb2, (short)0, acc[1][2], false, false);
    acc[1][3] = __builtin_amdgcn_wmma_f32_16x16x32_bf16(false, fa1, false, fb3, (short)0, acc[1][3], false, false);

    __syncthreads();                 // everyone done reading this tile
    storeA(); storeB();
    __syncthreads();                 // next tile visible
    k0 = kn;
  }

  // final tile (no prefetch)
  {
    v16bf fa0 = load_frag_a(lA, wm + 0,  lane);
    v16bf fa1 = load_frag_a(lA, wm + 16, lane);
    v16bf fb0 = load_frag_b(lB, wn + 0,  lane);
    v16bf fb1 = load_frag_b(lB, wn + 16, lane);
    v16bf fb2 = load_frag_b(lB, wn + 32, lane);
    v16bf fb3 = load_frag_b(lB, wn + 48, lane);

    acc[0][0] = __builtin_amdgcn_wmma_f32_16x16x32_bf16(false, fa0, false, fb0, (short)0, acc[0][0], false, false);
    acc[0][1] = __builtin_amdgcn_wmma_f32_16x16x32_bf16(false, fa0, false, fb1, (short)0, acc[0][1], false, false);
    acc[0][2] = __builtin_amdgcn_wmma_f32_16x16x32_bf16(false, fa0, false, fb2, (short)0, acc[0][2], false, false);
    acc[0][3] = __builtin_amdgcn_wmma_f32_16x16x32_bf16(false, fa0, false, fb3, (short)0, acc[0][3], false, false);
    acc[1][0] = __builtin_amdgcn_wmma_f32_16x16x32_bf16(false, fa1, false, fb0, (short)0, acc[1][0], false, false);
    acc[1][1] = __builtin_amdgcn_wmma_f32_16x16x32_bf16(false, fa1, false, fb1, (short)0, acc[1][1], false, false);
    acc[1][2] = __builtin_amdgcn_wmma_f32_16x16x32_bf16(false, fa1, false, fb2, (short)0, acc[1][2], false, false);
    acc[1][3] = __builtin_amdgcn_wmma_f32_16x16x32_bf16(false, fa1, false, fb3, (short)0, acc[1][3], false, false);
  }

  // ---- epilogue: store C ----
  const int half = lane >> 4, idx = lane & 15;
  if (fullMN) {
#pragma unroll
    for (int j = 0; j < 4; ++j) {
      const int colg = bn + wn + j * 16 + idx;
      const float bv = bias ? bias[colg] : 0.0f;
#pragma unroll
      for (int i = 0; i < 2; ++i) {
        const int rowBase = bm + wm + i * 16 + half * 8;
#pragma unroll
        for (int r = 0; r < 8; ++r) {
          float v = acc[i][j][r] + bv;
          if (relu) v = fmaxf(v, 0.0f);
          C[(long)(rowBase + r) * N + colg] = v;
        }
      }
    }
  } else {
#pragma unroll
    for (int j = 0; j < 4; ++j) {
      const int colg = bn + wn + j * 16 + idx;
      if (colg >= N) continue;
      const float bv = bias ? bias[colg] : 0.0f;
#pragma unroll
      for (int i = 0; i < 2; ++i) {
        const int rowBase = bm + wm + i * 16 + half * 8;
#pragma unroll
        for (int r = 0; r < 8; ++r) {
          const int rowg = rowBase + r;
          if (rowg < M) {
            float v = acc[i][j][r] + bv;
            if (relu) v = fmaxf(v, 0.0f);
            C[(long)rowg * N + colg] = v;
          }
        }
      }
    }
  }
}

// ---------------------------------------------------------------------------
// Fused per-feature kernel (one block per column d, 256 threads over N=1024):
//   pm = BN(mean_lin[:,d]); plv = BN(logvar_lin[:,d]);
//   z = pm + exp(0.5*plv)*eps -> out_z;  zx = BN(z) -> out_zx
// ---------------------------------------------------------------------------
__device__ inline void blk_reduce2_256(float& a, float& b, float* s1, float* s2) {
  const int t = threadIdx.x;
  s1[t] = a; s2[t] = b; __syncthreads();
  for (int off = 128; off > 0; off >>= 1) {
    if (t < off) { s1[t] += s1[t + off]; s2[t] += s2[t + off]; }
    __syncthreads();
  }
  a = s1[0]; b = s2[0]; __syncthreads();
}

__global__ __launch_bounds__(256) void fused_z_kernel(
    const float* __restrict__ mlin, const float* __restrict__ lvlin,
    const float* __restrict__ epsn, float* __restrict__ out_z,
    float* __restrict__ out_zx, int Nr, int D)
{
  __shared__ float s1[256], s2[256];
  const int d = blockIdx.x;
  const int t = threadIdx.x;
  const int RPT = 4;                 // Nr = 1024, 256 threads

  float mv[RPT], lv[RPT], zv[RPT];
  float sa = 0.f, sb = 0.f;
#pragma unroll
  for (int i = 0; i < RPT; ++i) {
    const int r = t + i * 256;
    const float x = mlin[(long)r * D + d];
    mv[i] = x; sa += x; sb += x * x;
  }
  blk_reduce2_256(sa, sb, s1, s2);
  const float m1 = sa / Nr, i1 = rsqrtf(sb / Nr - m1 * m1 + BN_EPS);

  sa = 0.f; sb = 0.f;
#pragma unroll
  for (int i = 0; i < RPT; ++i) {
    const int r = t + i * 256;
    const float x = lvlin[(long)r * D + d];
    lv[i] = x; sa += x; sb += x * x;
  }
  blk_reduce2_256(sa, sb, s1, s2);
  const float m2 = sa / Nr, i2 = rsqrtf(sb / Nr - m2 * m2 + BN_EPS);

  sa = 0.f; sb = 0.f;
#pragma unroll
  for (int i = 0; i < RPT; ++i) {
    const int r = t + i * 256;
    const float pm  = (mv[i] - m1) * i1;
    const float plv = (lv[i] - m2) * i2;
    const float z   = pm + __expf(0.5f * plv) * epsn[(long)r * D + d];
    zv[i] = z; sa += z; sb += z * z;
    out_z[(long)r * D + d] = z;
  }
  blk_reduce2_256(sa, sb, s1, s2);
  const float mz = sa / Nr, iz = rsqrtf(sb / Nr - mz * mz + BN_EPS);
#pragma unroll
  for (int i = 0; i < RPT; ++i) {
    const int r = t + i * 256;
    out_zx[(long)r * D + d] = (zv[i] - mz) * iz;
  }
}

// ---------------------------------------------------------------------------
// Column batchnorm: y[:,c] = BN(x[:,c]); one block per column.
// ---------------------------------------------------------------------------
__global__ __launch_bounds__(256) void bn_cols_kernel(
    const float* __restrict__ x, float* __restrict__ y, int R, int C)
{
  __shared__ float s1[256], s2[256];
  const int c = blockIdx.x;
  const int t = threadIdx.x;
  float sa = 0.f, sb = 0.f;
  for (int r = t; r < R; r += 256) {
    const float v = x[(long)r * C + c];
    sa += v; sb += v * v;
  }
  blk_reduce2_256(sa, sb, s1, s2);
  const float m = sa / R, inv = rsqrtf(sb / R - m * m + BN_EPS);
  for (int r = t; r < R; r += 256)
    y[(long)r * C + c] = (x[(long)r * C + c] - m) * inv;
}

// ---------------------------------------------------------------------------
// Row squared-norms: out[r] = sum_c x[r,c]^2   (128 threads)
// ---------------------------------------------------------------------------
__global__ __launch_bounds__(128) void rownorm_kernel(
    const float* __restrict__ x, float* __restrict__ out, int C)
{
  __shared__ float s[128];
  const int r = blockIdx.x, t = threadIdx.x;
  float a = 0.f;
  for (int c = t; c < C; c += 128) { const float v = x[(long)r * C + c]; a += v * v; }
  s[t] = a; __syncthreads();
  for (int off = 64; off > 0; off >>= 1) {
    if (t < off) s[t] += s[t + off];
    __syncthreads();
  }
  if (t == 0) out[r] = s[0];
}

// ---------------------------------------------------------------------------
// Gaussian-basis softmax over K=512 topics:
//   d[n,k] = -0.5*(|zx_n|^2 + |zc_k|^2 - 2*cross[n,k]);  phi = softmax_k(d)
// ---------------------------------------------------------------------------
__global__ __launch_bounds__(256) void dist_softmax_kernel(
    const float* __restrict__ cross, const float* __restrict__ zxn,
    const float* __restrict__ cn, float* __restrict__ phi, int Kc)
{
  __shared__ float s[256];
  const int n = blockIdx.x, t = threadIdx.x;
  const float zn = zxn[n];
  float dv[2];
  float mx = -3.0e38f;
#pragma unroll
  for (int i = 0; i < 2; ++i) {
    const int k = t + i * 256;
    const float d = -0.5f * (zn + cn[k] - 2.0f * cross[(long)n * Kc + k]);
    dv[i] = d; mx = fmaxf(mx, d);
  }
  s[t] = mx; __syncthreads();
  for (int off = 128; off > 0; off >>= 1) {
    if (t < off) s[t] = fmaxf(s[t], s[t + off]);
    __syncthreads();
  }
  mx = s[0]; __syncthreads();
  float se = 0.f;
#pragma unroll
  for (int i = 0; i < 2; ++i) { dv[i] = __expf(dv[i] - mx); se += dv[i]; }
  s[t] = se; __syncthreads();
  for (int off = 128; off > 0; off >>= 1) {
    if (t < off) s[t] += s[t + off];
    __syncthreads();
  }
  const float inv = 1.0f / s[0];
#pragma unroll
  for (int i = 0; i < 2; ++i) phi[(long)n * Kc + t + i * 256] = dv[i] * inv;
}

// ---------------------------------------------------------------------------
// In-place row softmax over wide rows (C = 30000): one block per row.
// ---------------------------------------------------------------------------
__global__ __launch_bounds__(256) void softmax_row_kernel(
    float* __restrict__ x, int C)
{
  __shared__ float s[256];
  const int r = blockIdx.x, t = threadIdx.x;
  float* row = x + (long)r * C;
  float mx = -3.0e38f;
  for (int c = t; c < C; c += 256) mx = fmaxf(mx, row[c]);
  s[t] = mx; __syncthreads();
  for (int off = 128; off > 0; off >>= 1) {
    if (t < off) s[t] = fmaxf(s[t], s[t + off]);
    __syncthreads();
  }
  mx = s[0]; __syncthreads();
  float se = 0.f;
  for (int c = t; c < C; c += 256) se += __expf(row[c] - mx);
  s[t] = se; __syncthreads();
  for (int off = 128; off > 0; off >>= 1) {
    if (t < off) s[t] += s[t + off];
    __syncthreads();
  }
  const float inv = 1.0f / s[0];
  for (int c = t; c < C; c += 256) row[c] = __expf(row[c] - mx) * inv;
}

// ---------------------------------------------------------------------------
// Launch
// ---------------------------------------------------------------------------
static inline void launch_gemm(const float* A, const float* B, const float* bias,
                               float* C, int M, int N, int K, int transB, int relu,
                               hipStream_t stream) {
  const dim3 grid((N + GBN - 1) / GBN, (M + GBM - 1) / GBM);
  if (transB)
    gemm_bf16_wmma<1><<<grid, NT, 0, stream>>>(A, B, bias, C, M, N, K, relu);
  else
    gemm_bf16_wmma<0><<<grid, NT, 0, stream>>>(A, B, bias, C, M, N, K, relu);
}

extern "C" void kernel_launch(void* const* d_in, const int* in_sizes, int n_in,
                              void* d_out, int out_size, void* d_ws, size_t ws_size,
                              hipStream_t stream) {
  const int N = 1024, V = 30000, EN1 = 1024, EN2 = 512, D = 128, K = 512, EMB = 300;

  const float* input_  = (const float*)d_in[0];
  // d_in[1] normalized_input_ : unused by reference outputs
  const float* W1      = (const float*)d_in[2];
  const float* b1      = (const float*)d_in[3];
  const float* W2      = (const float*)d_in[4];
  const float* b2      = (const float*)d_in[5];
  const float* Wmean   = (const float*)d_in[6];
  const float* bmean   = (const float*)d_in[7];
  const float* Wlogvar = (const float*)d_in[8];
  const float* blogvar = (const float*)d_in[9];
  const float* centres = (const float*)d_in[10];
  const float* mu_z    = (const float*)d_in[11];
  const float* emb     = (const float*)d_in[12];
  const float* epsn    = (const float*)d_in[13];

  // Output layout: (z, recon_v, zx, zc, zx_phi) concatenated
  float* out      = (float*)d_out;
  float* out_z    = out;
  float* out_rec  = out_z   + (long)N * D;
  float* out_zx   = out_rec + (long)N * V;
  float* out_zc   = out_zx  + (long)N * D;
  float* out_phi  = out_zc  + (long)K * D;

  // Workspace layout (floats); ~71 MB total
  float* ws    = (float*)d_ws;
  float* en1   = ws;                          // N*EN1
  float* en2   = en1  + (long)N * EN1;        // N*EN2
  float* mlin  = en2  + (long)N * EN2;        // N*D
  float* lvlin = mlin + (long)N * D;          // N*D
  float* beta  = lvlin + (long)N * D;         // K*V
  float* cross = beta + (long)K * V;          // N*K
  float* zxn   = cross + (long)N * K;         // N
  float* cn    = zxn + N;                     // K

  // encoder
  launch_gemm(input_, W1, b1, en1, N, EN1, V, 0, 1, stream);
  launch_gemm(en1, W2, b2, en2, N, EN2, EN1, 0, 1, stream);
  launch_gemm(en2, Wmean, bmean, mlin, N, D, EN2, 0, 0, stream);
  launch_gemm(en2, Wlogvar, blogvar, lvlin, N, D, EN2, 0, 0, stream);

  // BN(mean), BN(logvar), z, BN(z)  -> out_z, out_zx
  fused_z_kernel<<<D, 256, 0, stream>>>(mlin, lvlin, epsn, out_z, out_zx, N, D);

  // BN(centres) over topic axis -> out_zc
  bn_cols_kernel<<<D, 256, 0, stream>>>(centres, out_zc, K, D);

  // cross = zx @ zc^T  (WMMA, transB)
  launch_gemm(out_zx, out_zc, nullptr, cross, N, K, D, 1, 0, stream);
  rownorm_kernel<<<N, 128, 0, stream>>>(out_zx, zxn, D);
  rownorm_kernel<<<K, 128, 0, stream>>>(out_zc, cn, D);

  // zx_phi = softmax_k(-0.5 * ||zx - zc||^2)
  dist_softmax_kernel<<<N, 256, 0, stream>>>(cross, zxn, cn, out_phi, K);

  // beta = softmax_V(mu_z @ emb^T)
  launch_gemm(mu_z, emb, nullptr, beta, K, V, EMB, 1, 0, stream);
  softmax_row_kernel<<<K, 256, 0, stream>>>(beta, V);

  // recon_v = zx_phi @ beta
  launch_gemm(out_phi, beta, nullptr, out_rec, N, V, K, 0, 0, stream);

  (void)in_sizes; (void)n_in; (void)out_size; (void)ws_size;
}
